// MODEL_14663018348910
// MI455X (gfx1250) — matrix operand order repaired and checked by reading, not compiled, not run on hardware
//
#include <hip/hip_runtime.h>

// ---------------------------------------------------------------------------
// CDNA5 (gfx1250) ChebConv-LSTM: bf16 WMMA GEMMs with f32 accumulation.
// B=8, T=16, N=512, F=H=128, K=3 (Chebyshev order), 4H=512 gate width.
// Global->LDS tile staging uses GLOBAL_LOAD_ASYNC_TO_LDS_B128 (ASYNCcnt)
// so the DMA overlaps the WMMA stream instead of stalling on loadcnt.
// ---------------------------------------------------------------------------

typedef __bf16 bf16_t;
typedef __attribute__((ext_vector_type(16))) __bf16 v16bf;
typedef __attribute__((ext_vector_type(8)))  __bf16 v8bf;
typedef __attribute__((ext_vector_type(8)))  float  v8f;
typedef __attribute__((ext_vector_type(4)))  int    v4i;

// AS-qualified pointer typedefs (attribute binds to pointee via typedef).
typedef __attribute__((address_space(1))) v4i* gas_v4i_p;  // global
typedef __attribute__((address_space(3))) v4i* las_v4i_p;  // LDS (32-bit ptr)

#define LDSK 40  // padded LDS row stride (halfs) for a 32-wide K tile (16B-aligned chunks)

#if __has_builtin(__builtin_amdgcn_global_load_async_to_lds_b128)
#define HAVE_ASYNC_LDS 1
#endif

static __device__ __forceinline__ void wait_async_lds() {
#if defined(HAVE_ASYNC_LDS)
#if __has_builtin(__builtin_amdgcn_s_wait_asynccnt)
  __builtin_amdgcn_s_wait_asynccnt(0);
#else
  asm volatile("s_wait_asynccnt 0x0" ::: "memory");
#endif
#endif
}

static __device__ __forceinline__ bf16_t f2bf(float f) {
  unsigned u = __builtin_bit_cast(unsigned, f);
  unsigned r = (u + 0x7fffu + ((u >> 16) & 1u)) >> 16;  // round-to-nearest-even
  unsigned short s = (unsigned short)r;
  return __builtin_bit_cast(bf16_t, s);
}
static __device__ __forceinline__ float bf2f(bf16_t h) {
  unsigned short s = __builtin_bit_cast(unsigned short, h);
  unsigned u = ((unsigned)s) << 16;
  return __builtin_bit_cast(float, u);
}

// Stage a 128x32 bf16 tile (row-major, leading dim `ld`) from global into LDS.
static __device__ __forceinline__ void load_tile32(const bf16_t* __restrict__ g, int ld,
                                                   bf16_t* l, int tid) {
#pragma unroll
  for (int i = 0; i < 2; ++i) {
    int v  = tid + i * 256;   // 512 16-byte vectors total
    int r  = v >> 2;          // row 0..127
    int c8 = (v & 3) << 3;    // half-offset 0,8,16,24
#if defined(HAVE_ASYNC_LDS)
    // Direct memory->LDS DMA, tracked by ASYNCcnt; no VGPR staging, no
    // loadcnt stall between load and ds_store. AS1/AS3 pointers built via
    // integer round-trip (low 32 bits of a generic LDS pointer == LDS offset).
    __builtin_amdgcn_global_load_async_to_lds_b128(
        (gas_v4i_p)(size_t)(g + (size_t)r * ld + c8),
        (las_v4i_p)(unsigned)(size_t)(l + r * LDSK + c8),
        0, 0);
#else
    uint4 d = *(const uint4*)(g + (size_t)r * ld + c8);
    *(uint4*)(l + r * LDSK + c8) = d;
#endif
  }
}

// ---------------------------------------------------------------------------
// GEMM:  C[M,Nc] = A[M,K] * Bt[Nc,K]^T   (A row-major, Bt = B column-major)
// Block tile 128x128, 8 wave32s, each wave 32x64 (2x4 WMMA tiles), K step 32.
// MODE 1: out bf16 row-major + bf16 transposed          (T1 = L@X)
// MODE 2: out bf16 row-major of (2*acc - xsub)          (T2 = 2*L@T1 - X)
// MODE 3: out f32 of (acc + bias[n])                    (gate projection)
// All dims are multiples of the tile sizes -> no bounds checks.
// ---------------------------------------------------------------------------
template <int MODE>
__global__ __launch_bounds__(256) void k_gemm(
    const bf16_t* __restrict__ A,  long long sA, int lda,
    const bf16_t* __restrict__ Bt, long long sB, int ldb,
    int K,
    bf16_t* __restrict__ orm, long long sorm, int ldrm,
    bf16_t* __restrict__ otr, long long sotr, int ldtr,
    const bf16_t* __restrict__ xsub, long long sx, int ldx,
    const float* __restrict__ bias,
    float* __restrict__ of, long long sof, int ldf) {
  __shared__ __align__(16) bf16_t sm[2 * 2 * 128 * LDSK];

  const int tid  = threadIdx.x;
  const int lane = tid & 31;         // wave32
  const int wave = tid >> 5;         // 8 waves
  const int m0w  = (wave >> 1) * 32; // 4 waves along M
  const int n0w  = (wave & 1) * 64;  // 2 waves along N
  const int bz   = blockIdx.z;

  A  += (long long)bz * sA;
  Bt += (long long)bz * sB;
  const bf16_t* gA = A  + (size_t)(blockIdx.x * 128) * lda;
  const bf16_t* gB = Bt + (size_t)(blockIdx.y * 128) * ldb;

  v8f acc[2][4];
#pragma unroll
  for (int i = 0; i < 2; ++i)
#pragma unroll
    for (int j = 0; j < 4; ++j)
#pragma unroll
      for (int r = 0; r < 8; ++r) acc[i][j][r] = 0.0f;

  load_tile32(gA, lda, sm, tid);
  load_tile32(gB, ldb, sm + 128 * LDSK, tid);
  wait_async_lds();
  __syncthreads();

  const int nk = K >> 5;
  const int lr = lane & 15;
  const int ak = (lane >> 4) << 3;   // A frag: K chunk base 0 or 8 (+16 for upper half)
  const int bk = (lane >> 4) << 3;   // B frag: K chunk base 0 or 8 (+16 for upper half)

  for (int kt = 0; kt < nk; ++kt) {
    const int cur = kt & 1;
    bf16_t* lA = sm + cur * (2 * 128 * LDSK);
    bf16_t* lB = lA + 128 * LDSK;
    if (kt + 1 < nk) {
      // Kick off next tile's DMA; it lands in the other buffer while the
      // WMMAs below chew on the current one.
      bf16_t* nA = sm + (cur ^ 1) * (2 * 128 * LDSK);
      load_tile32(gA + (kt + 1) * 32, lda, nA, tid);
      load_tile32(gB + (kt + 1) * 32, ldb, nA + 128 * LDSK, tid);
    }

    // A fragments: lane l holds row M=(l&15); K halves {ak..ak+7} and {ak+16..ak+23}
    v16bf af[2];
#pragma unroll
    for (int mi = 0; mi < 2; ++mi) {
      const bf16_t* p = lA + (m0w + mi * 16 + lr) * LDSK;
      v8bf lo = *(const v8bf*)(p + ak);
      v8bf hi = *(const v8bf*)(p + ak + 16);
#pragma unroll
      for (int e = 0; e < 8; ++e) { af[mi][e] = lo[e]; af[mi][e + 8] = hi[e]; }
    }
    // B fragments: lane l holds col N=(l&15); K halves {bk..bk+7} and {bk+16..bk+23}
    v16bf bfr[4];
#pragma unroll
    for (int nj = 0; nj < 4; ++nj) {
      const bf16_t* p = lB + (n0w + nj * 16 + lr) * LDSK + bk;
      v8bf lo = *(const v8bf*)(p);
      v8bf hi = *(const v8bf*)(p + 16);
#pragma unroll
      for (int e = 0; e < 8; ++e) { bfr[nj][e] = lo[e]; bfr[nj][e + 8] = hi[e]; }
    }
#pragma unroll
    for (int mi = 0; mi < 2; ++mi)
#pragma unroll
      for (int nj = 0; nj < 4; ++nj)
        acc[mi][nj] = __builtin_amdgcn_wmma_f32_16x16x32_bf16(
            false, af[mi], false, bfr[nj], (short)0, acc[mi][nj], false, false);

    wait_async_lds();   // drain this wave's async stores before the barrier
    __syncthreads();
  }

  // Epilogue. C/D layout: lane 0-15 -> N=lane, rows M=r; lane 16-31 -> rows M=r+8.
  const int hb = (lane >> 4) << 3;
  const int mB = blockIdx.x * 128 + m0w;
  const int nB = blockIdx.y * 128 + n0w;

  if (MODE == 3) {
    float* ofp = of + (long long)bz * sof;
#pragma unroll
    for (int mi = 0; mi < 2; ++mi)
#pragma unroll
      for (int nj = 0; nj < 4; ++nj) {
        const int n  = nB + nj * 16 + lr;
        const int mb = mB + mi * 16 + hb;
        const float bv = bias[n];
#pragma unroll
        for (int r = 0; r < 8; ++r)
          ofp[(size_t)(mb + r) * ldf + n] = acc[mi][nj][r] + bv;
      }
  } else if (MODE == 1) {
    bf16_t* ormp = orm + (long long)bz * sorm;
    bf16_t* otrp = otr + (long long)bz * sotr;
#pragma unroll
    for (int mi = 0; mi < 2; ++mi)
#pragma unroll
      for (int nj = 0; nj < 4; ++nj) {
        const int n  = nB + nj * 16 + lr;
        const int mb = mB + mi * 16 + hb;
        v8bf pk;
#pragma unroll
        for (int r = 0; r < 8; ++r) {
          bf16_t h = f2bf(acc[mi][nj][r]);
          ormp[(size_t)(mb + r) * ldrm + n] = h;
          pk[r] = h;
        }
        *(v8bf*)(otrp + (size_t)n * ldtr + mb) = pk;  // contiguous transposed store
      }
  } else {  // MODE 2
    bf16_t* ormp = orm + (long long)bz * sorm;
    const bf16_t* xp = xsub + (long long)bz * sx;
#pragma unroll
    for (int mi = 0; mi < 2; ++mi)
#pragma unroll
      for (int nj = 0; nj < 4; ++nj) {
        const int n  = nB + nj * 16 + lr;
        const int mb = mB + mi * 16 + hb;
#pragma unroll
        for (int r = 0; r < 8; ++r) {
          float xv = bf2f(xp[(size_t)(mb + r) * ldx + n]);
          ormp[(size_t)(mb + r) * ldrm + n] = f2bf(2.0f * acc[mi][nj][r] - xv);
        }
      }
  }
}

// ---------------------------------------------------------------------------
// Elementwise kernels
// ---------------------------------------------------------------------------
__global__ void k_zero(unsigned* __restrict__ p, int n) {
  int i = blockIdx.x * 256 + threadIdx.x;
  if (i < n) p[i] = 0u;
}

__global__ void k_f2bf(const float* __restrict__ in, bf16_t* __restrict__ out, int n) {
  int i = blockIdx.x * 256 + threadIdx.x;
  if (i < n) out[i] = f2bf(in[i]);
}

// Build Wcat^T [512 out, 768 k] bf16 from W1[3,128,512], W2[3,128,512] (f32).
__global__ void k_wcat(const float* __restrict__ W1, const float* __restrict__ W2,
                       bf16_t* __restrict__ Wc) {
  int i = blockIdx.x * 256 + threadIdx.x;
  if (i >= 512 * 768) return;
  int o = i / 768, k = i - o * 768;
  int kk = k >> 7, f = k & 127;
  const float* W = (kk < 3) ? W1 : W2;
  int k3 = (kk < 3) ? kk : kk - 3;
  Wc[(size_t)o * 768 + k] = f2bf(W[((size_t)k3 * 128 + f) * 512 + o]);
}

__global__ void k_biassum(const float* __restrict__ b1, const float* __restrict__ b2,
                          float* __restrict__ bs) {
  int i = blockIdx.x * 256 + threadIdx.x;
  if (i < 512) bs[i] = b1[i] + b2[i];
}

// x[b,t,n,f] (f32) -> Acat0 slot0 (bf16 row-major, ld 768) + Xt0 (bf16 [b,f,n]).
__global__ void k_xconv(const float* __restrict__ x, int t,
                        bf16_t* __restrict__ acat, bf16_t* __restrict__ xt) {
  int i = blockIdx.x * 256 + threadIdx.x;  // B*N*F = 524288
  if (i >= 8 * 512 * 128) return;
  int b = i >> 16, n = (i >> 7) & 511, f = i & 127;
  bf16_t v = f2bf(x[(((size_t)(b * 16 + t) * 512 + n) << 7) + f]);
  acat[((size_t)(b * 512 + n)) * 768 + f] = v;
  xt[((size_t)(b * 128 + f)) * 512 + n]   = v;
}

// LSTM gates: Z[row,512]=[i|f|o|g]; updates c; scatters h where needed.
__global__ void k_gate(const float* __restrict__ Z, float* __restrict__ c,
                       bf16_t* __restrict__ acat_self,   // h -> own Acat col 384
                       bf16_t* __restrict__ ht_self,     // h transposed [b,f,n]
                       bf16_t* __restrict__ acat_next,   // h -> next layer Acat col 0 (or null)
                       float* __restrict__ outSeq, int t,// layer-1 output (or null)
                       float* __restrict__ hT, float* __restrict__ cT) {
  int i = blockIdx.x * 256 + threadIdx.x;  // B*N*H = 524288
  if (i >= 8 * 512 * 128) return;
  int b = i >> 16, n = (i >> 7) & 511, j = i & 127;
  size_t row = (size_t)(b * 512 + n);
  const float* z = Z + row * 512;
  float zi = z[j], zf = z[128 + j], zo = z[256 + j], zg = z[384 + j];
  float si = 1.0f / (1.0f + __expf(-zi));
  float sf = 1.0f / (1.0f + __expf(-zf));
  float so = 1.0f / (1.0f + __expf(-zo));
  float cn = sf * c[i] + si * tanhf(zg);
  float hn = so * tanhf(cn);
  c[i] = cn;
  bf16_t hb = f2bf(hn);
  acat_self[row * 768 + 384 + j] = hb;
  ht_self[((size_t)(b * 128 + j)) * 512 + n] = hb;
  if (acat_next) acat_next[row * 768 + j] = hb;
  if (outSeq) outSeq[(((size_t)(b * 16 + t) * 512 + n) << 7) + j] = hn;
  if (hT) { hT[i] = hn; cT[i] = cn; }
}

// ---------------------------------------------------------------------------
// Launcher
// ---------------------------------------------------------------------------
extern "C" void kernel_launch(void* const* d_in, const int* in_sizes, int n_in,
                              void* d_out, int out_size, void* d_ws, size_t ws_size,
                              hipStream_t stream) {
  (void)in_sizes; (void)n_in; (void)out_size; (void)ws_size;
  const float* x     = (const float*)d_in[0];
  const float* graph = (const float*)d_in[1];
  const float* W1_0  = (const float*)d_in[2];
  const float* b1_0  = (const float*)d_in[3];
  const float* W2_0  = (const float*)d_in[4];
  const float* b2_0  = (const float*)d_in[5];
  const float* W1_1  = (const float*)d_in[6];
  const float* b1_1  = (const float*)d_in[7];
  const float* W2_1  = (const float*)d_in[8];
  const float* b2_1  = (const float*)d_in[9];

  float* outSeq = (float*)d_out;                          // [B,T,N,H]
  float* hT = outSeq + (size_t)8 * 16 * 512 * 128;        // [B,N,H]
  float* cT = hT + (size_t)8 * 512 * 128;                 // [B,N,H]

  char* wsp = (char*)d_ws;
  size_t off = 0;
  auto alloc = [&](size_t bytes) -> void* {
    size_t a = (bytes + 255) & ~(size_t)255;
    void* p = wsp + off; off += a; return p;
  };

  bf16_t* Lbf   = (bf16_t*)alloc((size_t)8 * 512 * 512 * 2);  // graph bf16
  bf16_t* Wc0   = (bf16_t*)alloc((size_t)512 * 768 * 2);
  bf16_t* Wc1   = (bf16_t*)alloc((size_t)512 * 768 * 2);
  float*  bias0 = (float*)alloc(512 * 4);
  float*  bias1 = (float*)alloc(512 * 4);
  bf16_t* Acat0 = (bf16_t*)alloc((size_t)8 * 512 * 768 * 2);  // [x|T1x|T2x|h|T1h|T2h]
  bf16_t* Acat1 = (bf16_t*)alloc((size_t)8 * 512 * 768 * 2);
  bf16_t* Xt0   = (bf16_t*)alloc((size_t)8 * 128 * 512 * 2);  // x^T per batch
  bf16_t* T1t0  = (bf16_t*)alloc((size_t)8 * 128 * 512 * 2);
  bf16_t* T1t1  = (bf16_t*)alloc((size_t)8 * 128 * 512 * 2);
  bf16_t* T1ht0 = (bf16_t*)alloc((size_t)8 * 128 * 512 * 2);
  bf16_t* T1ht1 = (bf16_t*)alloc((size_t)8 * 128 * 512 * 2);
  bf16_t* ht0   = (bf16_t*)alloc((size_t)8 * 128 * 512 * 2);  // h0^T (also layer-1 x^T)
  bf16_t* ht1   = (bf16_t*)alloc((size_t)8 * 128 * 512 * 2);
  float*  Zb    = (float*)alloc((size_t)8 * 512 * 512 * 4);   // shared gate pre-activations
  float*  c0    = (float*)alloc((size_t)8 * 512 * 128 * 4);
  float*  c1    = (float*)alloc((size_t)8 * 512 * 128 * 4);

  // Zero recurrent state (graph-capture-safe kernels, no memset).
  k_zero<<<6144, 256, 0, stream>>>((unsigned*)Acat0, 8 * 512 * 768 / 2);
  k_zero<<<6144, 256, 0, stream>>>((unsigned*)Acat1, 8 * 512 * 768 / 2);
  k_zero<<<1024, 256, 0, stream>>>((unsigned*)ht0, 8 * 128 * 512 / 2);
  k_zero<<<1024, 256, 0, stream>>>((unsigned*)ht1, 8 * 128 * 512 / 2);
  k_zero<<<2048, 256, 0, stream>>>((unsigned*)c0, 8 * 512 * 128);
  k_zero<<<2048, 256, 0, stream>>>((unsigned*)c1, 8 * 512 * 128);

  // One-time conversions.
  k_f2bf<<<8192, 256, 0, stream>>>(graph, Lbf, 8 * 512 * 512);
  k_wcat<<<1536, 256, 0, stream>>>(W1_0, W2_0, Wc0);
  k_wcat<<<1536, 256, 0, stream>>>(W1_1, W2_1, Wc1);
  k_biassum<<<2, 256, 0, stream>>>(b1_0, b2_0, bias0);
  k_biassum<<<2, 256, 0, stream>>>(b1_1, b2_1, bias1);

  const long long sL  = (long long)512 * 512;   // L batch stride
  const long long sT  = (long long)128 * 512;   // transposed buffer batch stride
  const long long sAc = (long long)512 * 768;   // Acat batch stride

  auto prop1 = [&](const bf16_t* Btr, bf16_t* acat_slot, bf16_t* ttr) {
    k_gemm<1><<<dim3(4, 1, 8), 256, 0, stream>>>(
        Lbf, sL, 512, Btr, sT, 512, 512,
        acat_slot, sAc, 768, ttr, sT, 512,
        nullptr, 0, 0, nullptr, nullptr, 0, 0);
  };
  auto prop2 = [&](const bf16_t* Btr, bf16_t* acat_slot, const bf16_t* xs) {
    k_gemm<2><<<dim3(4, 1, 8), 256, 0, stream>>>(
        Lbf, sL, 512, Btr, sT, 512, 512,
        acat_slot, sAc, 768, nullptr, 0, 0,
        xs, sAc, 768, nullptr, nullptr, 0, 0);
  };
  auto zgemm = [&](const bf16_t* Acat, const bf16_t* Wc, const float* bs) {
    k_gemm<3><<<dim3(32, 4, 1), 256, 0, stream>>>(
        Acat, 0, 768, Wc, 0, 768, 768,
        nullptr, 0, 0, nullptr, 0, 0, nullptr, 0, 0,
        bs, Zb, 0, 512);
  };

  for (int t = 0; t < 16; ++t) {
    // ---- layer 0 ----
    k_xconv<<<2048, 256, 0, stream>>>(x, t, Acat0, Xt0);
    prop1(Xt0,   Acat0 + 128, T1t0);            // T1x
    prop2(T1t0,  Acat0 + 256, Acat0);           // T2x = 2*L@T1x - x
    prop1(ht0,   Acat0 + 512, T1ht0);           // T1h
    prop2(T1ht0, Acat0 + 640, Acat0 + 384);     // T2h = 2*L@T1h - h
    zgemm(Acat0, Wc0, bias0);
    k_gate<<<2048, 256, 0, stream>>>(Zb, c0, Acat0, ht0, Acat1,
                                     nullptr, t, nullptr, nullptr);
    // ---- layer 1 (input = h0, already in Acat1 slot0 / ht0) ----
    prop1(ht0,   Acat1 + 128, T1t1);
    prop2(T1t1,  Acat1 + 256, Acat1);
    prop1(ht1,   Acat1 + 512, T1ht1);
    prop2(T1ht1, Acat1 + 640, Acat1 + 384);
    zgemm(Acat1, Wc1, bias1);
    k_gate<<<2048, 256, 0, stream>>>(Zb, c1, Acat1, ht1, nullptr,
                                     outSeq, t,
                                     (t == 15) ? hT : nullptr,
                                     (t == 15) ? cT : nullptr);
  }
}
